// ElmoLM_15676630630511
// MI455X (gfx1250) — matrix operand order repaired and checked by reading, not compiled, not run on hardware
//
#include <hip/hip_runtime.h>
#include <hip/hip_bf16.h>
#include <cstdint>
#include <cstddef>

// ---- problem sizes (ELMo reference) ----
#define B_   4
#define T_   512
#define TP_  514        // T + bos + eos
#define V_   32000
#define D_   512
#define C_   4096
#define G_   16384      // 4*C
#define L_   2
#define MROWS_ 2056     // B_*TP_
#define MPAD_  2064     // padded to 129 16-row tiles

typedef __attribute__((ext_vector_type(16))) __bf16 bf16x16;
typedef __attribute__((ext_vector_type(8)))  float  f32x8;

__device__ __forceinline__ float sigmoidf_(float x) { return 1.0f / (1.0f + __expf(-x)); }
__device__ __forceinline__ float clip3_(float x)    { return fminf(3.0f, fmaxf(-3.0f, x)); }

// ---------------------------------------------------------------------------
// Fragment loaders (wave32; A/B use contiguous k = 16h + s per lane half), vectorized,
// unconditional (A sources zero-padded to a multiple of 16 rows).
// ---------------------------------------------------------------------------
__device__ __forceinline__ bf16x16 load_a_bf16(const __bf16* __restrict__ A, int lda,
                                               int m0, int k0) {
  const int lane = threadIdx.x & 31;
  const int m = m0 + (lane & 15);
  const int h = lane >> 4;
  const __bf16* row = A + (size_t)m * lda + k0 + 16 * h;    // 16B aligned
  union { uint4 q[2]; bf16x16 v; } u;
  u.q[0] = *(const uint4*)(row);
  u.q[1] = *(const uint4*)(row + 8);
  return u.v;
}

// Packed-B: K x N bf16 stored as 32x16 fragment tiles ordered (nt*kTiles + kt);
// lane holds 32 contiguous bytes -> two b128; wave streams 1KB per k-tile.
__device__ __forceinline__ bf16x16 load_b_packed(const __bf16* __restrict__ P,
                                                 int kTiles, int kt, int nt) {
  const int lane = threadIdx.x & 31;
  const __bf16* p = P + (((size_t)nt * kTiles + kt) * 32 + lane) * 16;
  union { uint4 q[2]; bf16x16 v; } u;
  u.q[0] = *(const uint4*)(p);
  u.q[1] = *(const uint4*)(p + 8);
  return u.v;
}

// C/D 16x16 f32: lane l -> col n = l&15, half h = l>>4; VGPR i -> row m = 8h+i
__device__ __forceinline__ void store_c_f32(float* __restrict__ Dm, int ldd,
                                            int m0, int n0, int M, f32x8 c) {
  const int lane = threadIdx.x & 31;
  const int n = n0 + (lane & 15);
  const int h = lane >> 4;
#pragma unroll
  for (int i = 0; i < 8; ++i) {
    const int m = m0 + 8 * h + i;
    if (m < M) Dm[(size_t)m * ldd + n] = c[i];
  }
}

// ---------------------------------------------------------------------------
// One-time f32 -> packed-bf16 weight shuffle. srcTrans=0: src K x N row-major;
// srcTrans=1: src N x K row-major (embedding table E).
// ---------------------------------------------------------------------------
__global__ void k_pack_bf16(const float* __restrict__ src, __bf16* __restrict__ dst,
                            int K, int N, int srcTrans) {
  const size_t total = (size_t)K * N;
  const int kTiles = K >> 5;
  for (size_t o = (size_t)blockIdx.x * blockDim.x + threadIdx.x; o < total;
       o += (size_t)gridDim.x * blockDim.x) {
    const int r    = (int)(o & 15);
    const int lane = (int)((o >> 4) & 31);
    const size_t tile = o >> 9;
    const int kt = (int)(tile % kTiles);
    const int nt = (int)(tile / kTiles);
    const int k  = (kt << 5) + ((lane >> 4) << 4) + r;
    const int nn = (nt << 4) + (lane & 15);
    const float v = srcTrans ? src[(size_t)nn * K + k] : src[(size_t)k * N + nn];
    dst[o] = (__bf16)v;
  }
}

// ---------------------------------------------------------------------------
// Generic GEMM: D(MxN,f32) = A(padded,bf16,row-major) * Bpacked(KxN)
// One wave computes a 16x64 strip; software-pipelined: k-tile kt+1 fragments
// are loaded before consuming k-tile kt (distinct registers -> overlapped
// s_wait_loadcnt instead of full drains).
// ---------------------------------------------------------------------------
__global__ void k_wmma_gemm_bf16(const __bf16* __restrict__ A, const __bf16* __restrict__ Bp,
                                 float* __restrict__ Dm, int M, int N, int K,
                                 int lda, int ldd) {
  const int wave = threadIdx.x >> 5;
  const int nStrips = N >> 6;
  const int mTiles  = (M + 15) >> 4;
  const long long total = (long long)mTiles * nStrips;
  const long long strip = (long long)blockIdx.x * (blockDim.x >> 5) + wave;
  if (strip >= total) return;
  const int mt = (int)(strip / nStrips);
  const int ns = (int)(strip % nStrips);
  const int m0 = mt << 4;
  const int nt0 = ns << 2;
  const int kTiles = K >> 5;

  f32x8 acc0 = {0.f,0.f,0.f,0.f,0.f,0.f,0.f,0.f};
  f32x8 acc1 = acc0, acc2 = acc0, acc3 = acc0;

  bf16x16 ac  = load_a_bf16(A, lda, m0, 0);
  bf16x16 b0c = load_b_packed(Bp, kTiles, 0, nt0);
  bf16x16 b1c = load_b_packed(Bp, kTiles, 0, nt0 + 1);
  bf16x16 b2c = load_b_packed(Bp, kTiles, 0, nt0 + 2);
  bf16x16 b3c = load_b_packed(Bp, kTiles, 0, nt0 + 3);

  for (int kt = 0; kt < kTiles - 1; ++kt) {
    const int kn = kt + 1;
    bf16x16 an  = load_a_bf16(A, lda, m0, kn << 5);
    bf16x16 b0n = load_b_packed(Bp, kTiles, kn, nt0);
    bf16x16 b1n = load_b_packed(Bp, kTiles, kn, nt0 + 1);
    bf16x16 b2n = load_b_packed(Bp, kTiles, kn, nt0 + 2);
    bf16x16 b3n = load_b_packed(Bp, kTiles, kn, nt0 + 3);
    acc0 = __builtin_amdgcn_wmma_f32_16x16x32_bf16(false, ac, false, b0c, (short)0, acc0, false, false);
    acc1 = __builtin_amdgcn_wmma_f32_16x16x32_bf16(false, ac, false, b1c, (short)0, acc1, false, false);
    acc2 = __builtin_amdgcn_wmma_f32_16x16x32_bf16(false, ac, false, b2c, (short)0, acc2, false, false);
    acc3 = __builtin_amdgcn_wmma_f32_16x16x32_bf16(false, ac, false, b3c, (short)0, acc3, false, false);
    ac = an; b0c = b0n; b1c = b1n; b2c = b2n; b3c = b3n;
  }
  acc0 = __builtin_amdgcn_wmma_f32_16x16x32_bf16(false, ac, false, b0c, (short)0, acc0, false, false);
  acc1 = __builtin_amdgcn_wmma_f32_16x16x32_bf16(false, ac, false, b1c, (short)0, acc1, false, false);
  acc2 = __builtin_amdgcn_wmma_f32_16x16x32_bf16(false, ac, false, b2c, (short)0, acc2, false, false);
  acc3 = __builtin_amdgcn_wmma_f32_16x16x32_bf16(false, ac, false, b3c, (short)0, acc3, false, false);

  store_c_f32(Dm, ldd, m0, (nt0 << 4),      M, acc0);
  store_c_f32(Dm, ldd, m0, (nt0 << 4) + 16, M, acc1);
  store_c_f32(Dm, ldd, m0, (nt0 << 4) + 32, M, acc2);
  store_c_f32(Dm, ldd, m0, (nt0 << 4) + 48, M, acc3);
}

// ---------------------------------------------------------------------------
// LSTM recurrence step 1: gates = Xin[t] + h_{t-1} @ W_state + bias; cell
// update; hpre = sigmoid(o)*tanh(c_new). One wave: 16-col slice of C across
// all 4 gates (shared A fragment), software-pipelined K loop.
// grid = (32, 2 dirs), block 256. h buffer 16 rows/dir (rows 4..15 zero).
// ---------------------------------------------------------------------------
__global__ void k_lstm_gate(const float* __restrict__ XinF, const float* __restrict__ XinB,
                            const __bf16* __restrict__ Ws, const float* __restrict__ bias,
                            int layer, int step,
                            const __bf16* __restrict__ h_bf, float* __restrict__ c_f,
                            __bf16* __restrict__ hpre) {
  const int dir = blockIdx.y;
  const int t   = dir ? (TP_ - 1 - step) : step;
  const float*  Xin = dir ? XinB : XinF;
  const __bf16* Wsl = Ws   + (size_t)(dir * L_ + layer) * D_ * G_;
  const float*  bl  = bias + (size_t)(dir * L_ + layer) * G_;
  const __bf16* h   = h_bf + (size_t)dir * 16 * D_;

  const int wave = threadIdx.x >> 5;
  const int tile = blockIdx.x * (blockDim.x >> 5) + wave;   // 0..255 (= C/16)
  const int lane = threadIdx.x & 31;
  const int nl   = (tile << 4) + (lane & 15);
  const int hh   = lane >> 4;
  const int kTiles = D_ >> 5;                               // 16

  f32x8 acc[4];
#pragma unroll
  for (int gq = 0; gq < 4; ++gq) {
    const int col = gq * C_ + nl;
#pragma unroll
    for (int i = 0; i < 8; ++i) {
      const int m = 8 * hh + i;
      acc[gq][i] = (m < B_) ? (Xin[((size_t)m * TP_ + t) * G_ + col] + bl[col]) : 0.0f;
    }
  }

  bf16x16 ac  = load_a_bf16(h, D_, 0, 0);
  bf16x16 b0c = load_b_packed(Wsl, kTiles, 0, tile);
  bf16x16 b1c = load_b_packed(Wsl, kTiles, 0, 256 + tile);
  bf16x16 b2c = load_b_packed(Wsl, kTiles, 0, 512 + tile);
  bf16x16 b3c = load_b_packed(Wsl, kTiles, 0, 768 + tile);

  for (int kt = 0; kt < kTiles - 1; ++kt) {
    const int kn = kt + 1;
    bf16x16 an  = load_a_bf16(h, D_, 0, kn << 5);
    bf16x16 b0n = load_b_packed(Wsl, kTiles, kn, tile);
    bf16x16 b1n = load_b_packed(Wsl, kTiles, kn, 256 + tile);
    bf16x16 b2n = load_b_packed(Wsl, kTiles, kn, 512 + tile);
    bf16x16 b3n = load_b_packed(Wsl, kTiles, kn, 768 + tile);
    acc[0] = __builtin_amdgcn_wmma_f32_16x16x32_bf16(false, ac, false, b0c, (short)0, acc[0], false, false);
    acc[1] = __builtin_amdgcn_wmma_f32_16x16x32_bf16(false, ac, false, b1c, (short)0, acc[1], false, false);
    acc[2] = __builtin_amdgcn_wmma_f32_16x16x32_bf16(false, ac, false, b2c, (short)0, acc[2], false, false);
    acc[3] = __builtin_amdgcn_wmma_f32_16x16x32_bf16(false, ac, false, b3c, (short)0, acc[3], false, false);
    ac = an; b0c = b0n; b1c = b1n; b2c = b2n; b3c = b3n;
  }
  acc[0] = __builtin_amdgcn_wmma_f32_16x16x32_bf16(false, ac, false, b0c, (short)0, acc[0], false, false);
  acc[1] = __builtin_amdgcn_wmma_f32_16x16x32_bf16(false, ac, false, b1c, (short)0, acc[1], false, false);
  acc[2] = __builtin_amdgcn_wmma_f32_16x16x32_bf16(false, ac, false, b2c, (short)0, acc[2], false, false);
  acc[3] = __builtin_amdgcn_wmma_f32_16x16x32_bf16(false, ac, false, b3c, (short)0, acc[3], false, false);

#pragma unroll
  for (int i = 0; i < 8; ++i) {
    const int m = 8 * hh + i;
    if (m < B_) {
      const float iv = sigmoidf_(acc[0][i]);
      const float fv = sigmoidf_(acc[1][i]);
      const float gv = tanhf(acc[2][i]);
      const float ov = sigmoidf_(acc[3][i]);
      const size_t cidx = (size_t)(dir * B_ + m) * C_ + nl;
      const float cn = clip3_(iv * gv + fv * c_f[cidx]);
      c_f[cidx] = cn;
      hpre[(size_t)(dir * 16 + m) * C_ + nl] = (__bf16)(ov * tanhf(cn));
    }
  }
}

// ---------------------------------------------------------------------------
// LSTM recurrence step 2: h_new = clip(hpre @ W_proj); write bf16 h (padded)
// + residual-added sequence output (f32 + bf16). Software-pipelined K loop.
// grid = (4, 2 dirs), block 256. W_proj packed: K=C (128 kt), N=D.
// ---------------------------------------------------------------------------
__global__ void k_lstm_proj(const __bf16* __restrict__ Wp, const __bf16* __restrict__ hpre,
                            int layer, int step, int addRes,
                            __bf16* __restrict__ h_bf,
                            const float* __restrict__ seqInF, const float* __restrict__ seqInB,
                            float* __restrict__ outFf, float* __restrict__ outBf,
                            __bf16* __restrict__ outFb, __bf16* __restrict__ outBb) {
  const int dir = blockIdx.y;
  const int t   = dir ? (TP_ - 1 - step) : step;
  const __bf16* Wpl = Wp + (size_t)(dir * L_ + layer) * C_ * D_;
  const __bf16* A   = hpre + (size_t)dir * 16 * C_;
  const float*  seqIn = dir ? seqInB : seqInF;
  float*  outF = dir ? outBf : outFf;
  __bf16* outX = dir ? outBb : outFb;

  const int wave = threadIdx.x >> 5;
  const int tile = blockIdx.x * (blockDim.x >> 5) + wave;   // 0..31 (= D/16)
  const int lane = threadIdx.x & 31;
  const int nl   = (tile << 4) + (lane & 15);
  const int hh   = lane >> 4;
  const int kTiles = C_ >> 5;                               // 128

  f32x8 acc = {0.f,0.f,0.f,0.f,0.f,0.f,0.f,0.f};
  bf16x16 ac = load_a_bf16(A, C_, 0, 0);
  bf16x16 bc = load_b_packed(Wpl, kTiles, 0, tile);
  for (int kt = 0; kt < kTiles - 1; ++kt) {
    const int kn = kt + 1;
    bf16x16 an = load_a_bf16(A, C_, 0, kn << 5);
    bf16x16 bn = load_b_packed(Wpl, kTiles, kn, tile);
    acc = __builtin_amdgcn_wmma_f32_16x16x32_bf16(false, ac, false, bc, (short)0, acc, false, false);
    ac = an; bc = bn;
  }
  acc = __builtin_amdgcn_wmma_f32_16x16x32_bf16(false, ac, false, bc, (short)0, acc, false, false);

#pragma unroll
  for (int i = 0; i < 8; ++i) {
    const int m = 8 * hh + i;
    if (m < B_) {
      const float hn = clip3_(acc[i]);
      h_bf[(size_t)(dir * 16 + m) * D_ + nl] = (__bf16)hn;
      const size_t sidx = ((size_t)m * TP_ + t) * D_ + nl;
      const float ov = hn + (addRes ? seqIn[sidx] : 0.0f);
      outF[sidx] = ov;
      outX[sidx] = (__bf16)ov;
    }
  }
}

// ---------------------------------------------------------------------------
// Small helper kernels
// ---------------------------------------------------------------------------
__global__ void k_zero_u32(uint32_t* __restrict__ p, size_t n) {
  for (size_t i = (size_t)blockIdx.x * blockDim.x + threadIdx.x; i < n;
       i += (size_t)gridDim.x * blockDim.x)
    p[i] = 0u;
}

__global__ void k_embed(const int* __restrict__ words, const float* __restrict__ E,
                        const float* __restrict__ bos, const float* __restrict__ eos,
                        float* __restrict__ xf, __bf16* __restrict__ xb) {
  const size_t n = (size_t)B_ * TP_ * D_;
  for (size_t i = (size_t)blockIdx.x * blockDim.x + threadIdx.x; i < n;
       i += (size_t)gridDim.x * blockDim.x) {
    const int d = (int)(i % D_);
    const size_t r = i / D_;
    const int t = (int)(r % TP_);
    const int b = (int)(r / TP_);
    float v;
    if (t == 0)            v = bos[d];
    else if (t == TP_ - 1) v = eos[d];
    else                   v = E[(size_t)words[b * T_ + (t - 1)] * D_ + d];
    xf[i] = v;
    xb[i] = (__bf16)v;
  }
}

__global__ void k_mix(const float* __restrict__ xf,
                      const float* __restrict__ l1f, const float* __restrict__ l1b,
                      const float* __restrict__ l2f, const float* __restrict__ l2b,
                      const float* __restrict__ s, const float* __restrict__ gamma,
                      float* __restrict__ out) {
  const float e0 = __expf(s[0]), e1 = __expf(s[1]), e2 = __expf(s[2]);
  const float inv = 1.0f / (e0 + e1 + e2);
  const float w0 = e0 * inv, w1 = e1 * inv, w2 = e2 * inv, gm = gamma[0];
  const size_t n = (size_t)B_ * T_ * 2 * D_;
  for (size_t i = (size_t)blockIdx.x * blockDim.x + threadIdx.x; i < n;
       i += (size_t)gridDim.x * blockDim.x) {
    const int d2 = (int)(i % (2 * D_));
    const size_t r = i / (2 * D_);
    const int tt = (int)(r % T_);
    const int b  = (int)(r / T_);
    const int t = tt + 1;
    const int d = d2 & (D_ - 1);
    const bool lo = d2 < D_;
    const size_t sidx = ((size_t)b * TP_ + t) * D_ + d;
    const float v0 = xf[sidx];
    const float v1 = lo ? l1f[sidx] : l1b[sidx];
    const float v2 = lo ? l2f[sidx] : l2b[sidx];
    out[i] = gm * (w0 * v0 + w1 * v1 + w2 * v2);
  }
}

__global__ void k_mask(int* __restrict__ p) {
  const int n = B_ * T_;
  for (int i = blockIdx.x * blockDim.x + threadIdx.x; i < n; i += gridDim.x * blockDim.x)
    p[i] = 1;
}

// tied = concat(E,E) ->  logit = (out_lo + out_hi) @ E^T ; build S = lo+hi (bf16)
__global__ void k_make_s(const float* __restrict__ out, __bf16* __restrict__ S) {
  const size_t n = (size_t)B_ * T_ * D_;
  for (size_t i = (size_t)blockIdx.x * blockDim.x + threadIdx.x; i < n;
       i += (size_t)gridDim.x * blockDim.x) {
    const int d = (int)(i % D_);
    const size_t r = i / D_;   // b*T + t
    S[i] = (__bf16)(out[r * (2 * D_) + d] + out[r * (2 * D_) + D_ + d]);
  }
}

// ---------------------------------------------------------------------------
// Host launcher
// ---------------------------------------------------------------------------
extern "C" void kernel_launch(void* const* d_in, const int* in_sizes, int n_in,
                              void* d_out, int out_size, void* d_ws, size_t ws_size,
                              hipStream_t stream) {
  (void)in_sizes; (void)n_in; (void)out_size; (void)ws_size;

  const int*   words = (const int*)d_in[0];
  const float* Ein   = (const float*)d_in[1];
  const float* bos   = (const float*)d_in[2];
  const float* eos   = (const float*)d_in[3];
  const float* Wi    = (const float*)d_in[4];
  const float* Wst   = (const float*)d_in[5];
  const float* bia   = (const float*)d_in[6];
  const float* Wpj   = (const float*)d_in[7];
  const float* sv    = (const float*)d_in[8];
  const float* gam   = (const float*)d_in[9];

  char* ws = (char*)d_ws;
  size_t off = 0;
  auto alloc = [&](size_t bytes) -> char* {
    char* p = ws + off;
    off = (off + bytes + 255) & ~(size_t)255;
    return p;
  };

  const size_t nWi     = (size_t)2 * L_ * D_ * G_;
  const size_t nWp     = (size_t)2 * L_ * C_ * D_;
  const size_t nE      = (size_t)V_ * D_;
  const size_t nSeq    = (size_t)B_ * TP_ * D_;      // 2056 rows x 512
  const size_t nSeqPad = (size_t)MPAD_ * D_;         // 2064 rows x 512 (bf16 A)
  const size_t nXin    = (size_t)B_ * TP_ * G_;

  __bf16* Wi_bf  = (__bf16*)alloc(nWi * 2);
  __bf16* Ws_bf  = (__bf16*)alloc(nWi * 2);
  __bf16* Wp_bf  = (__bf16*)alloc(nWp * 2);
  __bf16* E_bf   = (__bf16*)alloc(nE * 2);
  float*  x_f    = (float*) alloc(nSeq * 4);
  __bf16* x_bf   = (__bf16*)alloc(nSeqPad * 2);
  float*  l1f_f  = (float*) alloc(nSeq * 4);
  float*  l1b_f  = (float*) alloc(nSeq * 4);
  __bf16* l1f_bf = (__bf16*)alloc(nSeqPad * 2);
  __bf16* l1b_bf = (__bf16*)alloc(nSeqPad * 2);
  float*  l2f_f  = (float*) alloc(nSeq * 4);
  float*  l2b_f  = (float*) alloc(nSeq * 4);
  __bf16* l2f_bf = (__bf16*)alloc(nSeqPad * 2);
  __bf16* l2b_bf = (__bf16*)alloc(nSeqPad * 2);
  float*  XinF   = (float*) alloc(nXin * 4);
  float*  XinB   = (float*) alloc(nXin * 4);
  __bf16* h_bf   = (__bf16*)alloc((size_t)2 * 16 * D_ * 2);   // 16 rows/dir
  float*  c_f    = (float*) alloc((size_t)2 * B_ * C_ * 4);
  __bf16* hpre   = (__bf16*)alloc((size_t)2 * 16 * C_ * 2);   // 16 rows/dir
  __bf16* S_bf   = (__bf16*)alloc((size_t)B_ * T_ * D_ * 2);  // 2048 rows

  float* outPtr   = (float*)d_out;
  int*   maskPtr  = (int*)(outPtr + (size_t)B_ * T_ * 2 * D_);
  float* logitPtr = outPtr + (size_t)B_ * T_ * 2 * D_ + (size_t)B_ * T_;

  // 0) zero padded regions once per call
  const size_t padElems = ((size_t)(MPAD_ - MROWS_)) * D_;
  k_zero_u32<<<4, 256, 0, stream>>>((uint32_t*)(x_bf   + nSeq), padElems / 2);
  k_zero_u32<<<4, 256, 0, stream>>>((uint32_t*)(l1f_bf + nSeq), padElems / 2);
  k_zero_u32<<<4, 256, 0, stream>>>((uint32_t*)(l1b_bf + nSeq), padElems / 2);
  k_zero_u32<<<64, 256, 0, stream>>>((uint32_t*)hpre, (size_t)2 * 16 * C_ / 2);

  // 1) weight / embedding conversion + fragment packing
  for (int s = 0; s < 2 * L_; ++s) {
    k_pack_bf16<<<2048, 256, 0, stream>>>(Wi  + (size_t)s * D_ * G_,
                                          Wi_bf + (size_t)s * D_ * G_, D_, G_, 0);
    k_pack_bf16<<<2048, 256, 0, stream>>>(Wst + (size_t)s * D_ * G_,
                                          Ws_bf + (size_t)s * D_ * G_, D_, G_, 0);
    k_pack_bf16<<<1024, 256, 0, stream>>>(Wpj + (size_t)s * C_ * D_,
                                          Wp_bf + (size_t)s * C_ * D_, C_, D_, 0);
  }
  k_pack_bf16<<<4096, 256, 0, stream>>>(Ein, E_bf, D_, V_, 1);

  // 2) embedding gather + bos/eos concat
  k_embed<<<1024, 256, 0, stream>>>(words, Ein, bos, eos, x_f, x_bf);

  auto gemm = [&](const __bf16* A, const __bf16* Bp, float* Dm, int M, int N, int K,
                  int lda, int ldd) {
    const long long strips = (long long)((M + 15) / 16) * (N / 64);
    const int blocks = (int)((strips + 7) / 8);
    k_wmma_gemm_bf16<<<blocks, 256, 0, stream>>>(A, Bp, Dm, M, N, K, lda, ldd);
  };

  // 3) two BiLSTM layers
  const __bf16* seqF_bf = x_bf;  const __bf16* seqB_bf = x_bf;
  const float*  seqF_f  = x_f;   const float*  seqB_f  = x_f;

  for (int layer = 0; layer < L_; ++layer) {
    gemm(seqF_bf, Wi_bf + (size_t)(0 * L_ + layer) * D_ * G_, XinF,
         MROWS_, G_, D_, D_, G_);
    gemm(seqB_bf, Wi_bf + (size_t)(1 * L_ + layer) * D_ * G_, XinB,
         MROWS_, G_, D_, D_, G_);

    k_zero_u32<<<8,   256, 0, stream>>>((uint32_t*)h_bf, (size_t)2 * 16 * D_ / 2);
    k_zero_u32<<<128, 256, 0, stream>>>((uint32_t*)c_f,  (size_t)2 * B_ * C_);

    float*  oFf = (layer == 0) ? l1f_f  : l2f_f;
    float*  oBf = (layer == 0) ? l1b_f  : l2b_f;
    __bf16* oFb = (layer == 0) ? l1f_bf : l2f_bf;
    __bf16* oBb = (layer == 0) ? l1b_bf : l2b_bf;
    const int addRes = (layer > 0) ? 1 : 0;

    for (int step = 0; step < TP_; ++step) {
      k_lstm_gate<<<dim3(32, 2), 256, 0, stream>>>(XinF, XinB, Ws_bf, bia, layer, step,
                                                   h_bf, c_f, hpre);
      k_lstm_proj<<<dim3(4, 2), 256, 0, stream>>>(Wp_bf, hpre, layer, step, addRes,
                                                  h_bf, seqF_f, seqB_f,
                                                  oFf, oBf, oFb, oBb);
    }
    seqF_bf = oFb; seqB_bf = oBb;
    seqF_f  = oFf; seqB_f  = oBf;
  }

  // 4) softmax-weighted layer mix -> output region of d_out
  k_mix<<<2048, 256, 0, stream>>>(x_f, l1f_f, l1b_f, l2f_f, l2b_f, sv, gam, outPtr);

  // 5) mask
  k_mask<<<8, 256, 0, stream>>>(maskPtr);

  // 6) tied-embedding logits: S = out_lo + out_hi ; logit = S @ E^T (packed)
  k_make_s<<<1024, 256, 0, stream>>>(outPtr, S_bf);
  gemm(S_bf, E_bf, logitPtr, B_ * T_, V_, D_, D_, V_);
}
